// RNNEncoder_670014898481
// MI455X (gfx1250) — matrix operand order repaired and checked by reading, not compiled, not run on hardware
//
#include <hip/hip_runtime.h>
#include <hip/hip_bf16.h>

// ---------------------------------------------------------------------------
// 2-layer bidirectional GRU (B=64, T=512, I=128, H=256) for gfx1250 (CDNA5).
// Strategy:
//   * bf16 WMMA (v_wmma_f32_16x16x32_bf16, f32 accum) for all matmuls.
//   * Phase GEMM: xg = x @ w_ih^T + b_ih over all timesteps (time-parallel).
//   * Phase SCAN: 8 persistent workgroups (4 batch-slices x 2 directions),
//     h state in LDS, per-step [16x256]x[256x768] WMMA GEMM + gate math.
//     A fragments hoisted to registers; xg register-prefetched at step top;
//     next step's xg lines warmed via global_prefetch.
//   * PyTorch weight layout [out,in] row-major == B^T row-major, which maps
//     to two per-lane 16B loads (k 8g.., 16+8g..) for the CDNA5 WMMA B fragment.
// ---------------------------------------------------------------------------

typedef __attribute__((ext_vector_type(16))) __bf16 v16bf;
typedef __attribute__((ext_vector_type(8)))  float  v8f;

#define GRU_B 64
#define GRU_T 512
#define GRU_I 128
#define GRU_H 256
#define GRU_M (GRU_B * GRU_T)   // 32768
#define GRU_G (3 * GRU_H)       // 768

__device__ __forceinline__ unsigned short f32_to_bf16(float f) {
    union { float f; unsigned u; } v; v.f = f;
    unsigned r = v.u + 0x7FFFu + ((v.u >> 16) & 1u);   // round-to-nearest-even
    return (unsigned short)(r >> 16);
}
__device__ __forceinline__ float bf16_to_f32(unsigned short h) {
    union { unsigned u; float f; } v; v.u = ((unsigned)h) << 16;
    return v.f;
}
__device__ __forceinline__ float sigmoidf_fast(float x) {
    return 1.0f / (1.0f + __expf(-x));
}

union FragBF { uint4 q[2]; v16bf v; };
union XReg   { uint4 q[2]; unsigned short s[16]; };

// ---------------------------------------------------------------------------
// f32 -> bf16 conversion (4 elements / thread)
// ---------------------------------------------------------------------------
__global__ __launch_bounds__(256) void cvt_bf16_kernel(
    const float* __restrict__ src, unsigned short* __restrict__ dst, int n)
{
    int i = (blockIdx.x * 256 + threadIdx.x) * 4;
    if (i + 3 < n) {
        float4 f = *(const float4*)(src + i);
        dst[i + 0] = f32_to_bf16(f.x);
        dst[i + 1] = f32_to_bf16(f.y);
        dst[i + 2] = f32_to_bf16(f.z);
        dst[i + 3] = f32_to_bf16(f.w);
    } else {
        for (; i < n; ++i) dst[i] = f32_to_bf16(src[i]);
    }
}

// ---------------------------------------------------------------------------
// xg[d, m, n] = sum_k xbf[m,k] * wbf[d, n, k] + bias[d, n]     (bf16 out)
//   xbf : [M, K] bf16 row-major (activations)
//   wbf : [2, 768, K] bf16 row-major (== B^T, i.e. column-major B)
// Grid: (M/128, 768/64, 2), block 256 (8 waves); each wave: 16M x 64N tile.
// ---------------------------------------------------------------------------
__global__ __launch_bounds__(256) void gemm_xg_kernel(
    const unsigned short* __restrict__ xbf,
    const unsigned short* __restrict__ wbf,
    const float* __restrict__ bias,
    unsigned short* __restrict__ xg,
    int M, int K)
{
    const int lane = threadIdx.x & 31;
    const int wave = threadIdx.x >> 5;
    const int g    = lane >> 4;        // lane group 0/1
    const int mr   = lane & 15;
    const int d    = blockIdx.z;
    const int m0   = blockIdx.x * 128 + wave * 16;
    const int n0   = blockIdx.y * 64;

    const unsigned short* __restrict__ W    = wbf + (size_t)d * GRU_G * K;
    const unsigned short* __restrict__ Arow = xbf + (size_t)(m0 + mr) * K;

    v8f acc0 = {}, acc1 = {}, acc2 = {}, acc3 = {};
    for (int k0 = 0; k0 < K; k0 += 32) {
        FragBF a;
        a.q[0] = *(const uint4*)(Arow + k0 + 8 * g);
        a.q[1] = *(const uint4*)(Arow + k0 + 16 + 8 * g);
        FragBF b0, b1, b2, b3;
        const unsigned short* Bp = W + (size_t)(n0 + mr) * K + k0 + 8 * g;
        b0.q[0] = *(const uint4*)(Bp);
        b0.q[1] = *(const uint4*)(Bp + 16);
        Bp += (size_t)16 * K;
        b1.q[0] = *(const uint4*)(Bp);
        b1.q[1] = *(const uint4*)(Bp + 16);
        Bp += (size_t)16 * K;
        b2.q[0] = *(const uint4*)(Bp);
        b2.q[1] = *(const uint4*)(Bp + 16);
        Bp += (size_t)16 * K;
        b3.q[0] = *(const uint4*)(Bp);
        b3.q[1] = *(const uint4*)(Bp + 16);
        acc0 = __builtin_amdgcn_wmma_f32_16x16x32_bf16(false, a.v, false, b0.v, (short)0, acc0, false, false);
        acc1 = __builtin_amdgcn_wmma_f32_16x16x32_bf16(false, a.v, false, b1.v, (short)0, acc1, false, false);
        acc2 = __builtin_amdgcn_wmma_f32_16x16x32_bf16(false, a.v, false, b2.v, (short)0, acc2, false, false);
        acc3 = __builtin_amdgcn_wmma_f32_16x16x32_bf16(false, a.v, false, b3.v, (short)0, acc3, false, false);
    }

    v8f accs[4] = { acc0, acc1, acc2, acc3 };
#pragma unroll
    for (int nt = 0; nt < 4; ++nt) {
        const int n = n0 + nt * 16 + mr;              // C/D column = lane%16
        const float bv = bias[d * GRU_G + n];
        unsigned short* outp = xg + ((size_t)d * M + m0) * GRU_G + n;
#pragma unroll
        for (int v = 0; v < 8; ++v) {                 // C/D row = 8*g + v
            float val = accs[nt][v] + bv;
            outp[(size_t)(8 * g + v) * GRU_G] = f32_to_bf16(val);
        }
    }
}

// ---------------------------------------------------------------------------
// Recurrent scan: one workgroup = one (batch-slice of 16, direction) pair.
// Grid: 8 blocks x 256 threads.  LDS: h_bf 8KB + gh 48KB = 56KB.
// Per step:
//   1. register-prefetch xg[tt] (overlaps with GEMM), prefetch xg[tt+1] lines
//   2. gh = h_bf @ whh^T + b_hh: A fragments hoisted once from LDS,
//      48 tiles (8 waves x 6 tiles x 8 K-WMMAs), results to LDS
//   3. elementwise GRU gates in f32, h update, output store.
// ---------------------------------------------------------------------------
__global__ __launch_bounds__(256) void gru_scan_kernel(
    const unsigned short* __restrict__ xg,    // [2][M][768] bf16
    const unsigned short* __restrict__ whh,   // [2][768][256] bf16
    const float* __restrict__ bhh,            // [2][768] f32
    unsigned short* __restrict__ out_bf,      // [B][T][512] bf16 (layer0) or null
    float* __restrict__ out_f32)              // [B][T][512] f32 (layer1) or null
{
    __shared__ unsigned short h_bf[16 * GRU_H];  // current h, bf16, row-major
    __shared__ float          gh[16 * GRU_G];    // h @ whh^T + b_hh, f32

    const int d    = blockIdx.x & 1;       // direction
    const int s    = blockIdx.x >> 1;      // batch slice 0..3
    const int lane = threadIdx.x & 31;
    const int wave = threadIdx.x >> 5;
    const int g    = lane >> 4;
    const int mr   = lane & 15;

    // gate-phase mapping: thread -> (row gm, 16 consecutive cols at gj)
    const int gm = threadIdx.x >> 4;
    const int gj = (threadIdx.x & 15) * 16;

    const unsigned short* __restrict__ W    = whh + (size_t)d * GRU_G * GRU_H;
    const float* __restrict__          bias = bhh + d * GRU_G;
    const int base_b = s * 16;

    // xg row base for this thread's (batch row, time) — stride to walk time
    const unsigned short* __restrict__ xg_row0 =
        xg + ((size_t)d * GRU_M + (size_t)(base_b + gm) * GRU_T) * GRU_G;

    float h[16];
#pragma unroll
    for (int i = 0; i < 16; ++i) h[i] = 0.0f;
#pragma unroll
    for (int i = 0; i < 16; ++i) h_bf[gm * GRU_H + gj + i] = 0;
    __syncthreads();

    for (int t = 0; t < GRU_T; ++t) {
        const int tt = d ? (GRU_T - 1 - t) : t;

        // ---- Register-prefetch this step's xg (consumed after the GEMM) ----
        const unsigned short* xgrow = xg_row0 + (size_t)tt * GRU_G;
        XReg xr, xz, xn;
        xr.q[0] = *(const uint4*)(xgrow + gj);
        xr.q[1] = *(const uint4*)(xgrow + gj + 8);
        xz.q[0] = *(const uint4*)(xgrow + GRU_H + gj);
        xz.q[1] = *(const uint4*)(xgrow + GRU_H + gj + 8);
        xn.q[0] = *(const uint4*)(xgrow + 2 * GRU_H + gj);
        xn.q[1] = *(const uint4*)(xgrow + 2 * GRU_H + gj + 8);

        // ---- Warm next step's xg lines in the WGP cache ----
        if (t + 1 < GRU_T) {
            const int tn = d ? (GRU_T - 2 - t) : (t + 1);
            const unsigned short* xnext = xg_row0 + (size_t)tn * GRU_G;
            __builtin_prefetch(xnext + gj, 0, 3);
            __builtin_prefetch(xnext + GRU_H + gj, 0, 3);
            __builtin_prefetch(xnext + 2 * GRU_H + gj, 0, 3);
        }

        // ---- GEMM phase: hoist all 8 A fragments from LDS once ----
        const unsigned short* hrow = h_bf + mr * GRU_H;   // A row (lane's M)
        FragBF a[8];
#pragma unroll
        for (int kk = 0; kk < 8; ++kk) {
            a[kk].q[0] = *(const uint4*)(hrow + kk * 32 + 8 * g);
            a[kk].q[1] = *(const uint4*)(hrow + kk * 32 + 16 + 8 * g);
        }

        // wave handles N-tiles [6*wave, 6*wave+6)
        for (int q = 0; q < 6; ++q) {
            const int n = (wave * 6 + q) * 16 + mr;       // B column / D column
            const unsigned short* __restrict__ Brow = W + (size_t)n * GRU_H;
            v8f acc = {};
#pragma unroll
            for (int kk = 0; kk < 8; ++kk) {
                FragBF b;
                b.q[0] = *(const uint4*)(Brow + kk * 32 + 8 * g);
                b.q[1] = *(const uint4*)(Brow + kk * 32 + 16 + 8 * g);
                acc = __builtin_amdgcn_wmma_f32_16x16x32_bf16(
                          false, a[kk].v, false, b.v, (short)0, acc, false, false);
            }
            const float bv = bias[n];
#pragma unroll
            for (int v = 0; v < 8; ++v)
                gh[(8 * g + v) * GRU_G + n] = acc[v] + bv;
        }
        __syncthreads();   // gh complete before gate phase reads it

        // ---- Gate phase: row gm (batch b), cols gj..gj+15 of H ----
        const float* ghrow = gh + gm * GRU_G;
        unsigned short hb[16];
#pragma unroll
        for (int i = 0; i < 16; ++i) {
            const int j  = gj + i;
            float xrv = bf16_to_f32(xr.s[i]);
            float xzv = bf16_to_f32(xz.s[i]);
            float xnv = bf16_to_f32(xn.s[i]);
            float hr  = ghrow[j];
            float hz  = ghrow[GRU_H + j];
            float hn  = ghrow[2 * GRU_H + j];
            float r   = sigmoidf_fast(xrv + hr);
            float z   = sigmoidf_fast(xzv + hz);
            float nn  = tanhf(xnv + r * hn);
            h[i] = (1.0f - z) * nn + z * h[i];
            hb[i] = f32_to_bf16(h[i]);
        }
#pragma unroll
        for (int i = 0; i < 16; ++i) h_bf[gm * GRU_H + gj + i] = hb[i];

        // ---- Output store: [B][T][512], fwd cols 0..255, bwd 256..511 ----
        const size_t obase = ((size_t)(base_b + gm) * GRU_T + tt) * (2 * GRU_H)
                             + d * GRU_H + gj;
        if (out_bf) {
            unsigned short* o = out_bf + obase;
#pragma unroll
            for (int i = 0; i < 16; ++i) o[i] = hb[i];
        } else {
            float* o = out_f32 + obase;
#pragma unroll
            for (int i = 0; i < 16; ++i) o[i] = h[i];
        }
        __syncthreads();   // h_bf stable / gh reads done before next step
    }
}

// ---------------------------------------------------------------------------
// Host-side orchestration
// ---------------------------------------------------------------------------
static inline void launch_cvt(const float* src, unsigned short* dst, int n,
                              hipStream_t stream)
{
    int blocks = (n / 4 + 255) / 256;
    cvt_bf16_kernel<<<blocks, 256, 0, stream>>>(src, dst, n);
}

extern "C" void kernel_launch(void* const* d_in, const int* in_sizes, int n_in,
                              void* d_out, int out_size, void* d_ws, size_t ws_size,
                              hipStream_t stream)
{
    (void)in_sizes; (void)n_in; (void)out_size; (void)ws_size;

    const float* x    = (const float*)d_in[0];  // [64,512,128]
    const float* wih0 = (const float*)d_in[1];  // [2,768,128]
    const float* whh0 = (const float*)d_in[2];  // [2,768,256]
    const float* bih0 = (const float*)d_in[3];  // [2,768]
    const float* bhh0 = (const float*)d_in[4];  // [2,768]
    const float* wih1 = (const float*)d_in[5];  // [2,768,512]
    const float* whh1 = (const float*)d_in[6];  // [2,768,256]
    const float* bih1 = (const float*)d_in[7];  // [2,768]
    const float* bhh1 = (const float*)d_in[8];  // [2,768]
    float* out = (float*)d_out;                 // [64,512,512]

    // Workspace layout (all bf16/ushort), ~145 MB total:
    unsigned short* xbf0   = (unsigned short*)d_ws;                    // M*128
    unsigned short* xbf1   = xbf0   + (size_t)GRU_M * GRU_I;           // M*512
    unsigned short* wih_bf = xbf1   + (size_t)GRU_M * 512;             // 2*768*512
    unsigned short* whh_bf = wih_bf + (size_t)2 * GRU_G * 512;         // 2*768*256
    unsigned short* xg     = whh_bf + (size_t)2 * GRU_G * GRU_H;       // 2*M*768

    // ---- Layer 0 ----
    launch_cvt(x,    xbf0,   GRU_M * GRU_I,        stream);
    launch_cvt(wih0, wih_bf, 2 * GRU_G * GRU_I,    stream);
    launch_cvt(whh0, whh_bf, 2 * GRU_G * GRU_H,    stream);
    gemm_xg_kernel<<<dim3(GRU_M / 128, GRU_G / 64, 2), 256, 0, stream>>>(
        xbf0, wih_bf, bih0, xg, GRU_M, GRU_I);
    gru_scan_kernel<<<8, 256, 0, stream>>>(
        xg, whh_bf, bhh0, /*out_bf=*/xbf1, /*out_f32=*/nullptr);

    // ---- Layer 1 ----
    launch_cvt(wih1, wih_bf, 2 * GRU_G * 512,      stream);
    launch_cvt(whh1, whh_bf, 2 * GRU_G * GRU_H,    stream);
    gemm_xg_kernel<<<dim3(GRU_M / 128, GRU_G / 64, 2), 256, 0, stream>>>(
        xbf1, wih_bf, bih1, xg, GRU_M, 512);
    gru_scan_kernel<<<8, 256, 0, stream>>>(
        xg, whh_bf, bhh1, /*out_bf=*/nullptr, /*out_f32=*/out);
}
